// MambaSeqEmbedding_5102421148251
// MI455X (gfx1250) — compile-verified, matrix operand order checked by reading, not run on hardware
//
#include <hip/hip_runtime.h>
#include <hip/hip_bf16.h>

// ---------------------------------------------------------------------------
// Mamba (2-layer) forward for MI455X (gfx1250, wave32, WMMA).
// GEMMs: v_wmma_f32_16x16x32_f16, 32x64 wave tiles, ping-pong double-buffered
// fragment loads (no register-rotation moves; loads overlap WMMA).
// Scan: register-resident 16-state recurrence, chunked LDS staging of B/C.
// ---------------------------------------------------------------------------

#define D_MODEL 768
#define D_INNER 1536
#define D_STATE 16
#define D_CONV  4
#define DT_RANK 48
#define XD      (DT_RANK + 2 * D_STATE)   // 80
#define N_LAYERS 2
#define BATCH   2
#define SEQLEN  1024
#define MROWS   (BATCH * SEQLEN)          // 2048

typedef _Float16 half16 __attribute__((ext_vector_type(16)));
typedef _Float16 half8  __attribute__((ext_vector_type(8)));
typedef float    floatx8 __attribute__((ext_vector_type(8)));

// ---- WMMA fragment load -----------------------------------------------------
// 16-bit A-matrix 16x32 layout (ISA 7.12.2): lanes 0-15 hold row M=lane with
// K = {0..7} in VGPR0-3 and K = {16..23} in VGPR4-7; lanes 16-31 hold the same
// row's K = {8..15} and {24..31}. B (32x16, K-major weight row) mirrors this
// with lane = N column. Both load two contiguous 16B chunks per lane.
__device__ __forceinline__ half16 load_frag16(const _Float16* __restrict__ rowk, int sel) {
    half8 lo = *(const half8*)(rowk + sel * 8);
    half8 hi = *(const half8*)(rowk + 16 + sel * 8);
    return __builtin_shufflevector(lo, hi, 0, 1, 2, 3, 4, 5, 6, 7,
                                           8, 9, 10, 11, 12, 13, 14, 15);
}

__device__ __forceinline__ floatx8 wmma_f16(half16 a, half16 b, floatx8 c) {
    // (neg_a, A, neg_b, B, c_mod, C, reuse_a, reuse_b)
    return __builtin_amdgcn_wmma_f32_16x16x32_f16(false, a, false, b,
                                                  (short)0, c, false, false);
}

// ---- GEMM: C[M,N] = A[M,K] * W[N,K]^T (+ resid), f16 in / f32 out ----------
// One wave computes a (16*MT)(M) x (16*NT)(N) tile; 128-thread blocks = 4 waves.
// All tiling is compile-time exact: no guards, no divergence, EXEC stays all-1.
// Ping-pong double buffering: the two fragment sets alternate roles each
// 32-wide k-step, so there are no rotation copies and loads overlap WMMAs.
template <int N, int K, int MT, int NT, bool RESID>
__global__ void __launch_bounds__(128)
wmma_gemm_kernel(const _Float16* __restrict__ A, const _Float16* __restrict__ W,
                 const float* __restrict__ resid, float* __restrict__ C) {
    static_assert(N % (16 * NT) == 0, "N must tile exactly");
    static_assert(K % 64 == 0, "K must give an even number of 32-wide steps");
    constexpr int numN = N / (16 * NT);

    const int lane = threadIdx.x & 31;
    const int wave = threadIdx.x >> 5;
    const int lsub = lane & 15;      // M row (A) / N col (B) within subtile
    const int sel  = lane >> 4;      // K-half selector

    const int tile = blockIdx.x * 4 + wave;      // grid sized exactly
    const int tM = tile / numN;
    const int tN = tile - tM * numN;
    const int m0 = tM * (16 * MT);
    const int n0 = tN * (16 * NT);

    floatx8 c[MT][NT];
    #pragma unroll
    for (int i = 0; i < MT; ++i)
        #pragma unroll
        for (int j = 0; j < NT; ++j)
            #pragma unroll
            for (int r = 0; r < 8; ++r) c[i][j][r] = 0.0f;

    const _Float16* arow[MT];
    #pragma unroll
    for (int i = 0; i < MT; ++i) arow[i] = A + (size_t)(m0 + 16 * i + lsub) * K;
    const _Float16* brow = W + (size_t)(n0 + lsub) * K;

    half16 a0[MT], b0[NT], a1[MT], b1[NT];

    auto load_set = [&](half16 (&aa)[MT], half16 (&bb)[NT], int k0) {
        #pragma unroll
        for (int i = 0; i < MT; ++i) aa[i] = load_frag16(arow[i] + k0, sel);
        #pragma unroll
        for (int j = 0; j < NT; ++j)
            bb[j] = load_frag16(brow + (size_t)j * 16 * K + k0, sel);
    };
    auto do_wmma = [&](half16 (&aa)[MT], half16 (&bb)[NT]) {
        #pragma unroll
        for (int j = 0; j < NT; ++j)
            #pragma unroll
            for (int i = 0; i < MT; ++i)
                c[i][j] = wmma_f16(aa[i], bb[j], c[i][j]);
    };

    load_set(a0, b0, 0);
    #pragma unroll 1
    for (int k0 = 32; k0 < K - 32; k0 += 64) {
        load_set(a1, b1, k0);        // fly over set-0 WMMAs
        do_wmma(a0, b0);
        load_set(a0, b0, k0 + 32);   // fly over set-1 WMMAs
        do_wmma(a1, b1);
    }
    load_set(a1, b1, K - 32);
    do_wmma(a0, b0);
    do_wmma(a1, b1);

    // C/D layout: element (M = r + 8*sel, N = lsub) lives in VGPR r of lane.
    #pragma unroll
    for (int i = 0; i < MT; ++i)
        #pragma unroll
        for (int j = 0; j < NT; ++j) {
            const int col = n0 + j * 16 + lsub;
            #pragma unroll
            for (int r = 0; r < 8; ++r) {
                const int row = m0 + 16 * i + r + 8 * sel;
                const size_t idx = (size_t)row * N + col;
                float v = c[i][j][r];
                if constexpr (RESID) v += resid[idx];
                C[idx] = v;
            }
        }
}

// ---- RMSNorm (f32 in -> f16 out), one block per token row ------------------
__global__ void __launch_bounds__(256)
rmsnorm_f16_kernel(const float* __restrict__ h, const float* __restrict__ w,
                   _Float16* __restrict__ out) {
    __shared__ float red[256];
    const int m = blockIdx.x;
    const float* row = h + (size_t)m * D_MODEL;
    float s = 0.f;
    for (int j = threadIdx.x; j < D_MODEL; j += 256) { float v = row[j]; s += v * v; }
    red[threadIdx.x] = s;
    __syncthreads();
    for (int off = 128; off > 0; off >>= 1) {
        if (threadIdx.x < off) red[threadIdx.x] += red[threadIdx.x + off];
        __syncthreads();
    }
    const float r = rsqrtf(red[0] / (float)D_MODEL + 1e-5f);
    for (int j = threadIdx.x; j < D_MODEL; j += 256)
        out[(size_t)m * D_MODEL + j] = (_Float16)(row[j] * r * w[j]);
}

// ---- Final RMSNorm of the last token -> d_out (f32) ------------------------
__global__ void __launch_bounds__(256)
final_norm_kernel(const float* __restrict__ h, const float* __restrict__ w,
                  float* __restrict__ out) {
    __shared__ float red[256];
    const int b = blockIdx.x;
    const float* row = h + ((size_t)b * SEQLEN + (SEQLEN - 1)) * D_MODEL;
    float s = 0.f;
    for (int j = threadIdx.x; j < D_MODEL; j += 256) { float v = row[j]; s += v * v; }
    red[threadIdx.x] = s;
    __syncthreads();
    for (int off = 128; off > 0; off >>= 1) {
        if (threadIdx.x < off) red[threadIdx.x] += red[threadIdx.x + off];
        __syncthreads();
    }
    const float r = rsqrtf(red[0] / (float)D_MODEL + 1e-5f);
    for (int j = threadIdx.x; j < D_MODEL; j += 256)
        out[(size_t)b * D_MODEL + j] = row[j] * r * w[j];
}

// ---- f32 -> f16 elementwise convert (weights) ------------------------------
__global__ void __launch_bounds__(256)
cvt_f16_kernel(const float* __restrict__ src, _Float16* __restrict__ dst, int n) {
    const int i = blockIdx.x * 256 + threadIdx.x;
    if (i < n) dst[i] = (_Float16)src[i];
}

// ---- causal depthwise conv (taps=4) + bias + SiLU --------------------------
__global__ void __launch_bounds__(256)
conv_silu_kernel(const float* __restrict__ xz, const float* __restrict__ cw,
                 const float* __restrict__ cb, float* __restrict__ xcs,
                 _Float16* __restrict__ xcs16) {
    const int idx = blockIdx.x * 256 + threadIdx.x;      // < MROWS * D_INNER
    const int m = idx / D_INNER;
    const int d = idx - m * D_INNER;
    const int l = m & (SEQLEN - 1);
    float acc = cb[d];
    #pragma unroll
    for (int k = 0; k < D_CONV; ++k) {
        const int ls = l - (D_CONV - 1) + k;
        if (ls >= 0)
            acc += cw[d * D_CONV + k] * xz[(size_t)(m - (D_CONV - 1) + k) * (2 * D_INNER) + d];
    }
    const float v = acc / (1.f + __expf(-acc));          // SiLU
    xcs[idx] = v;
    xcs16[idx] = (_Float16)v;
}

// ---- dt_proj (K=48) + softplus (VALU; too small/awkward for WMMA) ----------
__global__ void __launch_bounds__(256)
dtproj_kernel(const float* __restrict__ xdbl, const float* __restrict__ dtw,
              const float* __restrict__ dtb, float* __restrict__ delta) {
    __shared__ float dtrow[DT_RANK];
    const int m = blockIdx.x;
    const int d = blockIdx.y * 256 + threadIdx.x;
    if (threadIdx.x < DT_RANK) dtrow[threadIdx.x] = xdbl[(size_t)m * XD + threadIdx.x];
    __syncthreads();
    float acc = dtb[d];
    #pragma unroll 8
    for (int r = 0; r < DT_RANK; ++r) acc += dtrow[r] * dtw[(size_t)d * DT_RANK + r];
    const float sp = acc > 0.f ? acc + log1pf(__expf(-acc)) : log1pf(__expf(acc));
    delta[(size_t)m * D_INNER + d] = sp;
}

// ---- selective scan + skip + SiLU(z) gate, emits f16 for out_proj ----------
// One thread owns channel d of batch b; 16-state recurrence lives in registers
// (16 independent FMA chains hide latency). B/C vectors staged in LDS in
// chunks of 8 timesteps (256 threads load exactly 8*32 floats per stage).
#define SCHUNK 8
__global__ void __launch_bounds__(256)
scan_kernel(const float* __restrict__ delta, const float* __restrict__ xcs,
            const float* __restrict__ xz, const float* __restrict__ xdbl,
            const float* __restrict__ A_log, const float* __restrict__ Dp,
            _Float16* __restrict__ yg16) {
    __shared__ float BmCm[SCHUNK][2 * D_STATE];
    const int chunksPerB = D_INNER / 256;                // 6
    const int b = blockIdx.x / chunksPerB;
    const int chunk = blockIdx.x - b * chunksPerB;
    const int d = chunk * 256 + threadIdx.x;

    float An[D_STATE];
    #pragma unroll
    for (int n = 0; n < D_STATE; ++n) An[n] = -__expf(A_log[(size_t)d * D_STATE + n]);
    const float Dv = Dp[d];

    float s[D_STATE];
    #pragma unroll
    for (int n = 0; n < D_STATE; ++n) s[n] = 0.f;

    const int tsub = threadIdx.x >> 5;                   // 0..7 : timestep in chunk
    const int nsub = threadIdx.x & 31;                   // 0..31: B/C element

    for (int l0 = 0; l0 < SEQLEN; l0 += SCHUNK) {
        __syncthreads();
        BmCm[tsub][nsub] =
            xdbl[((size_t)b * SEQLEN + l0 + tsub) * XD + DT_RANK + nsub];
        __syncthreads();

        // batch the per-chunk channel loads (independent of the recurrence)
        float dvv[SCHUNK], xvv[SCHUNK], zvv[SCHUNK];
        #pragma unroll
        for (int li = 0; li < SCHUNK; ++li) {
            const size_t m = (size_t)b * SEQLEN + l0 + li;
            dvv[li] = delta[m * D_INNER + d];
            xvv[li] = xcs[m * D_INNER + d];
            zvv[li] = xz[m * (2 * D_INNER) + D_INNER + d];
        }

        #pragma unroll
        for (int li = 0; li < SCHUNK; ++li) {
            const float dv = dvv[li];
            const float xv = xvv[li];
            const float zv = zvv[li];
            float y = 0.f;
            #pragma unroll
            for (int n = 0; n < D_STATE; ++n) {
                const float dA = __expf(dv * An[n]);
                s[n] = dA * s[n] + (dv * BmCm[li][n]) * xv;
                y += s[n] * BmCm[li][D_STATE + n];
            }
            const float yv = y + xv * Dv;
            const float g = zv / (1.f + __expf(-zv));    // SiLU(z)
            const size_t m = (size_t)b * SEQLEN + l0 + li;
            yg16[m * D_INNER + d] = (_Float16)(yv * g);
        }
    }
}

// ---------------------------------------------------------------------------
extern "C" void kernel_launch(void* const* d_in, const int* in_sizes, int n_in,
                              void* d_out, int out_size, void* d_ws, size_t ws_size,
                              hipStream_t stream) {
    const float* x       = (const float*)d_in[0];
    const float* in_w    = (const float*)d_in[1];   // [2, 3072, 768]
    const float* conv_w  = (const float*)d_in[2];   // [2, 1536, 4]
    const float* conv_b  = (const float*)d_in[3];   // [2, 1536]
    const float* xp_w    = (const float*)d_in[4];   // [2, 80, 1536]
    const float* dtp_w   = (const float*)d_in[5];   // [2, 1536, 48]
    const float* dtp_b   = (const float*)d_in[6];   // [2, 1536]
    const float* A_log   = (const float*)d_in[7];   // [2, 1536, 16]
    const float* D_par   = (const float*)d_in[8];   // [2, 1536]
    const float* out_w   = (const float*)d_in[9];   // [2, 768, 1536]
    const float* norm_w  = (const float*)d_in[10];  // [2, 768]
    const float* fnorm_w = (const float*)d_in[11];  // [768]
    float* out = (float*)d_out;                     // [2, 768]

    // ---- carve workspace (~81 MB) ----
    char* p = (char*)d_ws;
    auto carve = [&](size_t bytes) -> void* {
        void* r = (void*)p;
        p += (bytes + 255) & ~(size_t)255;
        return r;
    };
    float*    h      = (float*)   carve((size_t)MROWS * D_MODEL * 4);
    _Float16* hn16   = (_Float16*)carve((size_t)MROWS * D_MODEL * 2);
    _Float16* inw16  = (_Float16*)carve((size_t)2 * D_INNER * D_MODEL * 2);
    _Float16* xpw16  = (_Float16*)carve((size_t)XD * D_INNER * 2);
    _Float16* ow16   = (_Float16*)carve((size_t)D_MODEL * D_INNER * 2);
    float*    xz     = (float*)   carve((size_t)MROWS * 2 * D_INNER * 4);
    float*    xcs    = (float*)   carve((size_t)MROWS * D_INNER * 4);
    _Float16* xcs16  = (_Float16*)carve((size_t)MROWS * D_INNER * 2);
    float*    xdbl   = (float*)   carve((size_t)MROWS * XD * 4);
    float*    delta  = (float*)   carve((size_t)MROWS * D_INNER * 4);
    _Float16* yg16   = (_Float16*)carve((size_t)MROWS * D_INNER * 2);
    (void)ws_size; (void)in_sizes; (void)n_in; (void)out_size;

    // residual stream starts as a copy of x (don't mutate inputs)
    hipMemcpyAsync(h, x, (size_t)MROWS * D_MODEL * 4, hipMemcpyDeviceToDevice, stream);

    // wave-tile counts (all exact multiples of 4 waves per block):
    //   in_proj : (2048/32)*(3072/64) = 3072 tiles -> 768 blocks   (MT=2,NT=4)
    //   x_proj  : (2048/16)*(80/80)   =  128 tiles ->  32 blocks   (MT=1,NT=5)
    //   out_proj: (2048/32)*(768/64)  =  768 tiles -> 192 blocks   (MT=2,NT=4)

    for (int i = 0; i < N_LAYERS; ++i) {
        const float* inw_l = in_w   + (size_t)i * 2 * D_INNER * D_MODEL;
        const float* cw_l  = conv_w + (size_t)i * D_INNER * D_CONV;
        const float* cb_l  = conv_b + (size_t)i * D_INNER;
        const float* xpw_l = xp_w   + (size_t)i * XD * D_INNER;
        const float* dtw_l = dtp_w  + (size_t)i * D_INNER * DT_RANK;
        const float* dtb_l = dtp_b  + (size_t)i * D_INNER;
        const float* Al_l  = A_log  + (size_t)i * D_INNER * D_STATE;
        const float* Dp_l  = D_par  + (size_t)i * D_INNER;
        const float* ow_l  = out_w  + (size_t)i * D_MODEL * D_INNER;
        const float* nw_l  = norm_w + (size_t)i * D_MODEL;

        // 1) RMSNorm -> f16 activations
        rmsnorm_f16_kernel<<<MROWS, 256, 0, stream>>>(h, nw_l, hn16);

        // 2) weight converts for this layer
        {
            int n = 2 * D_INNER * D_MODEL;
            cvt_f16_kernel<<<(n + 255) / 256, 256, 0, stream>>>(inw_l, inw16, n);
            n = XD * D_INNER;
            cvt_f16_kernel<<<(n + 255) / 256, 256, 0, stream>>>(xpw_l, xpw16, n);
            n = D_MODEL * D_INNER;
            cvt_f16_kernel<<<(n + 255) / 256, 256, 0, stream>>>(ow_l, ow16, n);
        }

        // 3) in_proj: xz[2048,3072] = hn * inw^T   (WMMA, 32x64 tiles)
        wmma_gemm_kernel<2 * D_INNER, D_MODEL, 2, 4, false>
            <<<(MROWS / 32) * (2 * D_INNER / 64) / 4, 128, 0, stream>>>(
                hn16, inw16, nullptr, xz);

        // 4) causal depthwise conv + SiLU
        conv_silu_kernel<<<(MROWS * D_INNER) / 256, 256, 0, stream>>>(
            xz, cw_l, cb_l, xcs, xcs16);

        // 5) x_proj: xdbl[2048,80] = xc * xpw^T   (WMMA, NT=5 covers N=80 exactly)
        wmma_gemm_kernel<XD, D_INNER, 1, 5, false>
            <<<(MROWS / 16) / 4, 128, 0, stream>>>(
                xcs16, xpw16, nullptr, xdbl);

        // 6) dt_proj + softplus
        {
            dim3 g(MROWS, D_INNER / 256);
            dtproj_kernel<<<g, 256, 0, stream>>>(xdbl, dtw_l, dtb_l, delta);
        }

        // 7) selective scan + skip + SiLU(z) gate -> f16
        scan_kernel<<<BATCH * (D_INNER / 256), 256, 0, stream>>>(
            delta, xcs, xz, xdbl, Al_l, Dp_l, yg16);

        // 8) out_proj with fused residual add: h += yg * ow^T   (WMMA, 32x64)
        wmma_gemm_kernel<D_MODEL, D_INNER, 2, 4, true>
            <<<(MROWS / 32) * (D_MODEL / 64) / 4, 128, 0, stream>>>(
                yg16, ow16, h, h);
    }

    // final RMSNorm of last token per batch -> d_out [2,768]
    final_norm_kernel<<<BATCH, 256, 0, stream>>>(h, fnorm_w, out);
}
